// SelfAttention_23974507446344
// MI455X (gfx1250) — matrix operand order repaired and checked by reading, not compiled, not run on hardware
//
#include <hip/hip_runtime.h>
#include <hip/hip_bf16.h>

typedef _Float16 f16;
typedef __attribute__((ext_vector_type(16))) _Float16 v16h;
typedef __attribute__((ext_vector_type(8)))  _Float16 v8h;
typedef __attribute__((ext_vector_type(8)))  float    v8f;

#define Bc 2
#define Tc 2048
#define Cc 1024
#define Hc 16
#define Dc 64
#define BTc (Bc * Tc)

__device__ inline v8f wmma_f16(v16h a, v16h b, v8f c) {
    // 8 args: (neg_a, A, neg_b, B, c_mod, C, reuse_a, reuse_b)
    return __builtin_amdgcn_wmma_f32_16x16x32_f16(
        false, a, false, b, (short)0, c, false, false);
}

// ---- CDNA5 async global->LDS copy (16B per lane), tracked by ASYNCcnt ----
__device__ __forceinline__ void async_b128(void* lds_dst, const void* gbase,
                                           uint32_t byte_off) {
    uint32_t lds_addr = (uint32_t)(uintptr_t)lds_dst;   // low 32 bits = LDS offset
    uint64_t base = (uint64_t)(uintptr_t)gbase;         // uniform SGPR pair
    asm volatile("global_load_async_to_lds_b128 %0, %1, %2"
                 :: "v"(lds_addr), "v"(byte_off), "s"(base)
                 : "memory");
}
__device__ __forceinline__ void wait_async0() {
#if __has_builtin(__builtin_amdgcn_s_wait_asynccnt)
    __builtin_amdgcn_s_wait_asynccnt(0);
#else
    asm volatile("s_wait_asynccnt 0x0" ::: "memory");
#endif
}

// A fragment, 16x32 f16, layout: lane<16 -> row=lane, K=0..15;
// lane>=16 -> row=lane-16, K=16..31. src row-major, stride in f16.
__device__ inline v16h load_a_frag(const f16* base, int stride, int lane) {
    int half = lane >> 4, lr = lane & 15;
    const f16* row = base + (size_t)lr * stride;
    v16h r;
    ((v8h*)&r)[0] = *(const v8h*)(row + half * 16);
    ((v8h*)&r)[1] = *(const v8h*)(row + half * 16 + 8);
    return r;
}

// B fragment, 32x16 f16, from TRANSPOSED tile [N][K]:
// lane<16 -> N=lane, K=0..15 ; lane>=16 -> N=lane-16, K=16..31
__device__ inline v16h load_b_fragT(const f16* baseT, int stride, int lane) {
    int half = lane >> 4, lr = lane & 15;
    const f16* row = baseT + (size_t)lr * stride + half * 16;
    v16h r;
    ((v8h*)&r)[0] = *(const v8h*)(row);
    ((v8h*)&r)[1] = *(const v8h*)(row + 8);
    return r;
}

__global__ void cvt_f32_to_f16(const float* __restrict__ in,
                               f16* __restrict__ out, int n) {
    int i = blockIdx.x * blockDim.x + threadIdx.x;
    int i4 = i * 4;
    if (i4 + 3 < n) {
        float4 v = *(const float4*)(in + i4);
        f16 o[4] = {(f16)v.x, (f16)v.y, (f16)v.z, (f16)v.w};
        *(uint2*)(out + i4) = *(uint2*)o;
    } else {
        for (int k = i4; k < n && k < i4 + 4; ++k) out[k] = (f16)in[k];
    }
}

// in: [K][N] f32  ->  out: [N][K] f16 (one-time weight transpose+convert)
__global__ __launch_bounds__(256)
void transpose_cvt_f32_f16(const float* __restrict__ in, f16* __restrict__ out,
                           int K, int N) {
    __shared__ f16 tile[32][33];   // +1 pad vs bank conflicts
    int k0 = blockIdx.y * 32, n0 = blockIdx.x * 32;
    int tr = threadIdx.x >> 3;          // 0..31
    int tc = (threadIdx.x & 7) * 4;     // 0,4,...,28
    float4 v = *(const float4*)(in + (size_t)(k0 + tr) * N + n0 + tc);
    tile[tr][tc + 0] = (f16)v.x;
    tile[tr][tc + 1] = (f16)v.y;
    tile[tr][tc + 2] = (f16)v.z;
    tile[tr][tc + 3] = (f16)v.w;
    __syncthreads();
    f16 o[4] = {tile[tc + 0][tr], tile[tc + 1][tr],
                tile[tc + 2][tr], tile[tc + 3][tr]};
    *(uint2*)(out + (size_t)(n0 + tr) * K + k0 + tc) = *(uint2*)o;
}

// C = A[MxK] * Bt^T + bias, Bt given TRANSPOSED [N][K]; f16 in, f32 accum.
// Block tile 128x128, BK=32, 256 threads = 8 waves (4x2), wave tile 32x64.
// Double-buffered LDS; tiles arrive via async b128 copies (ASYNCcnt) that
// overlap with the WMMA phase of the previous tile.
template <bool OUT_F16>
__global__ __launch_bounds__(256)
void gemm_wmma_f16(const f16* __restrict__ A, const f16* __restrict__ Bt,
                   const float* __restrict__ bias, void* __restrict__ Cout,
                   int M, int N, int K) {
    constexpr int BM = 128, BN = 128, BK = 32;
    __shared__ f16 Al[2][BM][BK];   // row-major [M][K]   (16 KB)
    __shared__ f16 Bl[2][BN][BK];   // row-major [N][K]   (16 KB)

    int tid = threadIdx.x;
    int wave = tid >> 5, lane = tid & 31;
    int wm = wave >> 1, wn = wave & 1;
    int Mbase = blockIdx.y * BM, Nbase = blockIdx.x * BN;

    int row = tid >> 1, koff = (tid & 1) * 16;      // each thread: 16 f16 = 2 x b128
    uint32_t aoff = (uint32_t)(((size_t)(Mbase + row) * K + koff) * sizeof(f16));
    uint32_t boff = (uint32_t)(((size_t)(Nbase + row) * K + koff) * sizeof(f16));

    auto issue_tile = [&](int buf, int kt) {
        uint32_t kb = (uint32_t)(kt * sizeof(f16));
        async_b128(&Al[buf][row][koff],     A,  aoff + kb);
        async_b128(&Al[buf][row][koff + 8], A,  aoff + kb + 16);
        async_b128(&Bl[buf][row][koff],     Bt, boff + kb);
        async_b128(&Bl[buf][row][koff + 8], Bt, boff + kb + 16);
    };

    v8f acc[2][4] = {};

    issue_tile(0, 0);                 // prologue
    wait_async0();
    __syncthreads();

    int nk = K / BK;
    for (int it = 0; it < nk; ++it) {
        int cur = it & 1, nxt = cur ^ 1;
        if (it + 1 < nk) issue_tile(nxt, (it + 1) * BK);   // prefetch next tile

        v16h bf[4];
#pragma unroll
        for (int ni = 0; ni < 4; ++ni)
            bf[ni] = load_b_fragT(&Bl[cur][wn * 64 + ni * 16][0], BK, lane);
#pragma unroll
        for (int mi = 0; mi < 2; ++mi) {
            v16h af = load_a_frag(&Al[cur][wm * 32 + mi * 16][0], BK, lane);
#pragma unroll
            for (int ni = 0; ni < 4; ++ni)
                acc[mi][ni] = wmma_f16(af, bf[ni], acc[mi][ni]);
        }
        wait_async0();                 // next tile landed
        __syncthreads();               // everyone done with cur + nxt written
    }

    // f32 C/D layout: VGPR r: lanes0-15 -> M=r, N=lane; lanes16-31 -> M=8+r
    int half = lane >> 4, lr = lane & 15;
#pragma unroll
    for (int mi = 0; mi < 2; ++mi)
#pragma unroll
        for (int ni = 0; ni < 4; ++ni)
#pragma unroll
            for (int r = 0; r < 8; ++r) {
                int mg = Mbase + wm * 32 + mi * 16 + half * 8 + r;
                int ng = Nbase + wn * 64 + ni * 16 + lr;
                float v = acc[mi][ni][r] + bias[ng];
                if (OUT_F16)
                    ((f16*)Cout)[(size_t)mg * N + ng] = (f16)v;
                else
                    ((float*)Cout)[(size_t)mg * N + ng] = v;
            }
}

// Flash attention: causal, softmax in f32, matmuls via WMMA f16.
// Block: 256 thr = 8 waves, each wave a 16-row Q tile (block = 128 rows).
// Double-buffered K/V tiles: async K prefetch; next V held in registers
// through the compute phase, transposed into LDS afterwards.
__global__ __launch_bounds__(256)
void flash_attn_wmma(const f16* __restrict__ qkv, f16* __restrict__ attn_out) {
    constexpr int TJ = 32;
    const float scale = 0.125f;  // 1/sqrt(64)
    __shared__ f16 Kl[2][TJ][Dc];    // K rows            (8 KB)
    __shared__ f16 Vt[2][Dc][TJ];    // V transposed      (8 KB)
    __shared__ f16 Pl[8][16][TJ];    // per-wave P stage  (8 KB)

    int tid = threadIdx.x, wave = tid >> 5, lane = tid & 31;
    int half = lane >> 4, lr = lane & 15;
    int qblk = blockIdx.x, h = blockIdx.y, b = blockIdx.z;
    int q0 = qblk * 128 + wave * 16;

    const size_t C3 = (size_t)3 * Cc;
    const f16* qb = qkv + (size_t)b * Tc * C3 + h * Dc;
    const f16* vb = qb + 2 * Cc;

    // Q fragments, d-chunks {0..31},{32..63}, straight from global
    v16h qf[2];
#pragma unroll
    for (int dc = 0; dc < 2; ++dc) {
        const f16* rowp = qb + (size_t)(q0 + lr) * C3 + dc * 32;
        v16h r;
        ((v8h*)&r)[0] = *(const v8h*)(rowp + half * 16);
        ((v8h*)&r)[1] = *(const v8h*)(rowp + half * 16 + 8);
        qf[dc] = r;
    }

    v8f o[4] = {};
    float mrow[8], lrow[8];
#pragma unroll
    for (int r = 0; r < 8; ++r) { mrow[r] = -1e30f; lrow[r] = 0.0f; }

    int j = tid >> 3, d0 = (tid & 7) * 8;     // K/V tile loader mapping
    // K rows live at column offset C within qkv
    uint32_t kgoff = (uint32_t)((((size_t)b * Tc + j) * C3 + Cc + h * Dc + d0)
                                * sizeof(f16));
    uint32_t cstride = (uint32_t)(C3 * sizeof(f16));

    int njt = qblk * 4 + 4;           // j-tiles of 32, causal upper bound

    // prologue: tile 0
    async_b128(&Kl[0][j][d0], qkv, kgoff);
    {
        v8h vv = *(const v8h*)(vb + (size_t)j * C3 + d0);
#pragma unroll
        for (int i = 0; i < 8; ++i) Vt[0][d0 + i][j] = vv[i];
    }
    wait_async0();
    __syncthreads();

    for (int it = 0; it < njt; ++it) {
        int jc = it * TJ;
        int cur = it & 1, nxt = cur ^ 1;
        bool pf = (it + 1 < njt);
        v8h vv;
        if (pf) {   // prefetch: async K into spare buffer, V into registers
            int jc2 = jc + TJ;
            async_b128(&Kl[nxt][j][d0], qkv, kgoff + (uint32_t)jc2 * cstride);
            vv = *(const v8h*)(vb + (size_t)(jc2 + j) * C3 + d0);
        }

        float sv[2][8], mt[8];
#pragma unroll
        for (int r = 0; r < 8; ++r) mt[r] = -1e30f;

#pragma unroll
        for (int jh = 0; jh < 2; ++jh) {   // S = Q * K^T  (two 16x16 tiles)
            v8f s = {};
#pragma unroll
            for (int dc = 0; dc < 2; ++dc) {
                // B operand B[d][j]: lane<16 -> K[j0+lr][dc*32+0..15]
                const f16* rowp = &Kl[cur][jh * 16 + lr][dc * 32 + half * 16];
                v16h bf;
                ((v8h*)&bf)[0] = *(const v8h*)(rowp);
                ((v8h*)&bf)[1] = *(const v8h*)(rowp + 8);
                s = wmma_f16(qf[dc], bf, s);
            }
            int jg = jc + jh * 16 + lr;
#pragma unroll
            for (int r = 0; r < 8; ++r) {
                int qg = q0 + half * 8 + r;
                float v = s[r] * scale;
                if (jg > qg) v = -1e30f;   // causal mask
                sv[jh][r] = v;
                mt[r] = fmaxf(mt[r], v);
            }
        }
        // row max across the 16 lanes of each half (xor 1..8 stays in-half)
#pragma unroll
        for (int r = 0; r < 8; ++r) {
            float v = mt[r];
#pragma unroll
            for (int m = 1; m < 16; m <<= 1) v = fmaxf(v, __shfl_xor(v, m, 32));
            mt[r] = v;
        }
        float alpha[8];
#pragma unroll
        for (int r = 0; r < 8; ++r) {
            float mnew = fmaxf(mrow[r], mt[r]);
            alpha[r] = __expf(mrow[r] - mnew);
            mrow[r] = mnew;
        }
        // P = exp(S - m), stage to LDS in row-major for A-fragment reload
#pragma unroll
        for (int jh = 0; jh < 2; ++jh)
#pragma unroll
            for (int r = 0; r < 8; ++r) {
                float p = __expf(sv[jh][r] - mrow[r]);
                sv[jh][r] = p;
                Pl[wave][half * 8 + r][jh * 16 + lr] = (f16)p;
            }
#pragma unroll
        for (int r = 0; r < 8; ++r) {
            float ps = sv[0][r] + sv[1][r];
#pragma unroll
            for (int m = 1; m < 16; m <<= 1) ps += __shfl_xor(ps, m, 32);
            lrow[r] = lrow[r] * alpha[r] + ps;
#pragma unroll
            for (int dt = 0; dt < 4; ++dt) o[dt][r] *= alpha[r];
        }
        // O += P * V
        v16h pfrag = load_a_frag(&Pl[wave][0][0], TJ, lane);
#pragma unroll
        for (int dt = 0; dt < 4; ++dt) {
            v16h vf = load_b_fragT(&Vt[cur][dt * 16][0], TJ, lane);
            o[dt] = wmma_f16(pfrag, vf, o[dt]);
        }

        if (pf) {   // V transpose-store into spare buffer after compute
#pragma unroll
            for (int i = 0; i < 8; ++i) Vt[nxt][d0 + i][j] = vv[i];
        }
        wait_async0();
        __syncthreads();
    }

    // epilogue: O / l  ->  f16 [B*T][C] at head column
#pragma unroll
    for (int r = 0; r < 8; ++r) {
        float inv = 1.0f / lrow[r];
        int qg = q0 + half * 8 + r;
#pragma unroll
        for (int dt = 0; dt < 4; ++dt)
            attn_out[((size_t)b * Tc + qg) * Cc + h * Dc + dt * 16 + lr] =
                (f16)(o[dt][r] * inv);
    }
}

extern "C" void kernel_launch(void* const* d_in, const int* in_sizes, int n_in,
                              void* d_out, int out_size, void* d_ws, size_t ws_size,
                              hipStream_t stream) {
    const float* x      = (const float*)d_in[0];
    const float* w_attn = (const float*)d_in[1];
    const float* b_attn = (const float*)d_in[2];
    const float* w_proj = (const float*)d_in[3];
    const float* b_proj = (const float*)d_in[4];
    float* out = (float*)d_out;

    // f16 workspace layout
    f16* x_h    = (f16*)d_ws;                       // B*T*C
    f16* wa_t   = x_h  + (size_t)BTc * Cc;          // [3C][C] transposed
    f16* wp_t   = wa_t + (size_t)Cc * 3 * Cc;       // [C][C]  transposed
    f16* qkv_h  = wp_t + (size_t)Cc * Cc;           // B*T*3C
    f16* attn_h = qkv_h + (size_t)BTc * 3 * Cc;     // B*T*C

    {   // activations: plain convert
        int n = BTc * Cc, threads = 256, grid = (n / 4 + threads - 1) / threads;
        cvt_f32_to_f16<<<grid, threads, 0, stream>>>(x, x_h, n);
    }
    // weights: one-time transpose+convert -> [N][K] so GEMM B tiles are contiguous
    transpose_cvt_f32_f16<<<dim3(3 * Cc / 32, Cc / 32), 256, 0, stream>>>(
        w_attn, wa_t, Cc, 3 * Cc);
    transpose_cvt_f32_f16<<<dim3(Cc / 32, Cc / 32), 256, 0, stream>>>(
        w_proj, wp_t, Cc, Cc);

    // qkv = x @ w_attn + b_attn   (f16 out)
    gemm_wmma_f16<true><<<dim3(3 * Cc / 128, BTc / 128), 256, 0, stream>>>(
        x_h, wa_t, b_attn, (void*)qkv_h, BTc, 3 * Cc, Cc);

    // causal attention
    flash_attn_wmma<<<dim3(Tc / 128, Hc, Bc), 256, 0, stream>>>(qkv_h, attn_h);

    // out = attn @ w_proj + b_proj   (f32 out)
    gemm_wmma_f16<false><<<dim3(Cc / 128, BTc / 128), 256, 0, stream>>>(
        attn_h, wp_t, b_proj, (void*)out, BTc, Cc, Cc);
}
